// TransformerEncoder_17987323036070
// MI455X (gfx1250) — compile-verified, matrix operand order checked by reading, not run on hardware
//
#include <hip/hip_runtime.h>

typedef unsigned short u16;
typedef __bf16 bf16;
typedef bf16  v16bf __attribute__((ext_vector_type(16)));
typedef float v8f   __attribute__((ext_vector_type(8)));

struct __align__(16) U4 { unsigned x, y, z, w; };
union FragAB { U4 q[2]; v16bf v; };

// ---------------- constants (match reference shapes) ----------------
constexpr int NL = 4;          // layers
constexpr int NB = 8;          // batch (sequences)
constexpr int SQ = 512;        // new tokens per seq
constexpr int DM = 1024;       // model dim
constexpr int NH = 16;         // heads
constexpr int DH = 64;         // head dim
constexpr int PG = 16;         // page size
constexpr int NPAGES = 640;    // total pages
constexpr int CACHEDTOK = 768; // cached kv tokens per seq
constexpr int KVLEN = 1280;    // total kv per seq
constexpr int NT = NB * SQ;    // 4096 tokens
constexpr int FF = 4 * DM;     // 4096

// ---------------- helpers ----------------
__device__ __forceinline__ u16 f2bf(float f) {
  unsigned u = __float_as_uint(f);
  u += 0x7FFFu + ((u >> 16) & 1u);           // round-to-nearest-even
  return (u16)(u >> 16);
}

__device__ __forceinline__ v8f wmma_bf16(v16bf a, v16bf b, v8f c) {
  return __builtin_amdgcn_wmma_f32_16x16x32_bf16(false, a, false, b, (short)0, c,
                                                 false, false);
}

// A/B fragment: lane L: (M or N)=L%16, half=L/16 holds K in
// [k0+half*8, +8) and [k0+16+half*8, +8)  -> two 16B loads
__device__ __forceinline__ FragAB load_frag(const u16* rowp, int k0, int hh) {
  FragAB f;
  f.q[0] = *(const U4*)(rowp + k0 + hh * 8);
  f.q[1] = *(const U4*)(rowp + k0 + 16 + hh * 8);
  return f;
}

__device__ __forceinline__ float gelu_exact(float x) {
  return 0.5f * x * (1.0f + erff(x * 0.70710678118654752f));
}

// ---- CDNA5 async global->LDS copy (16B per lane), tracked by ASYNCcnt ----
// lds_off: low 32 bits of the flat shared address == LDS byte address.
__device__ __forceinline__ void async_ld_b128(unsigned lds_off, const void* gaddr) {
  asm volatile("global_load_async_to_lds_b128 %0, %1, off"
               :: "v"(lds_off), "v"(gaddr)
               : "memory");
}
__device__ __forceinline__ void wait_async0() {
  asm volatile("s_wait_asynccnt 0" ::: "memory");
}

// ---------------- weight transpose + cast  W[K,N] -> WT[N,K] bf16 --------------
__global__ __launch_bounds__(256)
void transpose_cast(const float* __restrict__ W, u16* __restrict__ WT, int K, int N) {
  size_t g = (size_t)blockIdx.x * 256 + threadIdx.x;
  if (g >= (size_t)K * N) return;
  int n = (int)(g / K), k = (int)(g % K);
  WT[g] = f2bf(W[(size_t)k * N + n]);
}

__global__ __launch_bounds__(256)
void cast_init(const float* __restrict__ x, float* __restrict__ xcur, u16* __restrict__ xbf) {
  int g = blockIdx.x * 256 + threadIdx.x;
  float v = x[g];
  xcur[g] = v;
  xbf[g]  = f2bf(v);
}

// ---------------- generic bf16 WMMA GEMM: out = act(A @ WT^T + bias)*scale -----
// Tiles staged into LDS with GLOBAL_LOAD_ASYNC_TO_LDS_B128 (ASYNCcnt path).
// MODE 0: fp32 out; MODE 1: bf16 out (scaled); MODE 2: bf16 out with exact GELU
template<int MODE>
__global__ __launch_bounds__(256)
void gemm_wmma_bf16(const u16* __restrict__ A, const u16* __restrict__ WT,
                    const float* __restrict__ bias, float* __restrict__ outF,
                    u16* __restrict__ outH, int M, int N, int K, float scale) {
  constexpr int LDT = 72;                     // padded LDS row stride (elements)
  __shared__ __align__(16) u16 lA[64 * LDT];
  __shared__ __align__(16) u16 lB[64 * LDT];
  const int tid  = threadIdx.x;
  const int lane = tid & 31;
  const int wv   = tid >> 5;
  const int ln16 = lane & 15;
  const int hh   = lane >> 4;
  const int nbase = blockIdx.x * 64;
  const int mbase = blockIdx.y * 64;
  const int wm = (wv >> 1) * 16;              // 4 wave-rows
  const int wn = (wv & 1) * 32;               // 2 wave-cols

  v8f acc0 = {}; v8f acc1 = {};

  const int r  = tid >> 2;                    // 0..63 tile row
  const int cc = (tid & 3) * 16;              // 16 bf16 per thread
  const u16* gA = A  + (size_t)(mbase + r) * K + cc;
  const u16* gB = WT + (size_t)(nbase + r) * K + cc;

  // LDS byte addresses for this thread's staging slots (2x16B in A, 2x16B in B)
  const unsigned aoff0 = (unsigned)(size_t)(lA + r * LDT + cc);
  const unsigned aoff1 = aoff0 + 16;
  const unsigned boff0 = (unsigned)(size_t)(lB + r * LDT + cc);
  const unsigned boff1 = boff0 + 16;

  for (int kb = 0; kb < K; kb += 64) {
    // async DMA: global -> LDS, no VGPR round trip
    async_ld_b128(aoff0, gA + kb);
    async_ld_b128(aoff1, gA + kb + 8);
    async_ld_b128(boff0, gB + kb);
    async_ld_b128(boff1, gB + kb + 8);
    if (kb + 64 < K) {                        // warm L2/GL1 for next tile
      __builtin_prefetch(gA + kb + 64, 0, 0);
      __builtin_prefetch(gB + kb + 64, 0, 0);
    }
    wait_async0();                            // own transfers landed in LDS
    __syncthreads();                          // publish to all waves
#pragma unroll
    for (int ks = 0; ks < 2; ++ks) {
      FragAB fa = load_frag(lA + (wm + ln16) * LDT, ks * 32, hh);
      FragAB f0 = load_frag(lB + (wn + ln16) * LDT, ks * 32, hh);
      FragAB f1 = load_frag(lB + (wn + 16 + ln16) * LDT, ks * 32, hh);
      acc0 = wmma_bf16(fa.v, f0.v, acc0);
      acc1 = wmma_bf16(fa.v, f1.v, acc1);
    }
    __syncthreads();                          // all reads done before next overwrite
  }

#pragma unroll
  for (int v = 0; v < 8; ++v) {
    int row  = mbase + wm + v + hh * 8;       // C layout: M = vgpr + half*8
    int col0 = nbase + wn + ln16;
    int col1 = col0 + 16;
    float x0 = acc0[v] + bias[col0];
    float x1 = acc1[v] + bias[col1];
    if (MODE == 2) { x0 = gelu_exact(x0); x1 = gelu_exact(x1); }
    x0 *= scale; x1 *= scale;
    if (MODE == 0) {
      outF[(size_t)row * N + col0] = x0;
      outF[(size_t)row * N + col1] = x1;
    } else {
      outH[(size_t)row * N + col0] = f2bf(x0);
      outH[(size_t)row * N + col1] = f2bf(x1);
    }
  }
}

// ---------------- gather paged KV + new KV into dense K[b,kv,D] / V^T[b,h,dh,kv]
__global__ __launch_bounds__(256)
void gather_kv(const float* __restrict__ pagetable, const int* __restrict__ indptr,
               const int* __restrict__ indices, const u16* __restrict__ knew,
               const u16* __restrict__ vnew, u16* __restrict__ Kf,
               u16* __restrict__ VT, int layer) {
  int g = blockIdx.x * 256 + threadIdx.x;
  if (g >= NB * KVLEN * DM) return;
  int c  = g & (DM - 1);
  int kv = (g / DM) % KVLEN;
  int b  = g / (DM * KVLEN);
  u16 kk, vv;
  if (kv < CACHEDTOK) {
    int page = indices[indptr[b] + (kv >> 4)];
    // pagetable[l, page, w, tok, h, dh] ; [tok][h*dh] contiguous as 1024
    size_t off = ((((size_t)layer * NPAGES + page) * 2) * PG + (kv & 15)) * DM + c;
    kk = f2bf(pagetable[off]);
    vv = f2bf(pagetable[off + (size_t)PG * DM]);   // w=1 plane
  } else {
    size_t t = (size_t)(b * SQ + (kv - CACHEDTOK)) * DM + c;
    kk = knew[t];
    vv = vnew[t];
  }
  Kf[(size_t)(b * KVLEN + kv) * DM + c] = kk;
  int h = c >> 6, dh = c & 63;
  VT[(((size_t)b * NH + h) * DH + dh) * (size_t)KVLEN + kv] = vv;
}

// ---------------- flash attention: per block = (b, h, 128 q-rows) ----------------
__global__ __launch_bounds__(256)
void attn_flash(const u16* __restrict__ Qs,   // pre-scaled bf16 [T][D]
                const u16* __restrict__ Kf,   // bf16 [B*KV][D]
                const u16* __restrict__ VT,   // bf16 [B*H*DH][KV]
                u16* __restrict__ Obf) {      // bf16 [T][D]
  __shared__ __align__(16) u16 lP[8 * 16 * 64];   // per-wave P tile (16x64)
  const int tid = threadIdx.x, lane = tid & 31, wv = tid >> 5;
  const int ln16 = lane & 15, hh = lane >> 4;
  const int idx = blockIdx.x;                 // ((b*H + h)*4 + qt)
  const int qt = idx & 3;
  const int h  = (idx >> 2) & 15;
  const int b  = idx >> 6;
  const int qrow0 = b * SQ + qt * 128 + wv * 16;

  const u16* qp = Qs + (size_t)(qrow0 + ln16) * DM + h * DH;
  FragAB aq0 = load_frag(qp, 0, hh);          // dh 0..31
  FragAB aq1 = load_frag(qp, 32, hh);         // dh 32..63

  float mi[8], li[8];
#pragma unroll
  for (int v = 0; v < 8; ++v) { mi[v] = -1e30f; li[v] = 0.0f; }
  v8f acc[4] = {};

  u16* myP = lP + wv * (16 * 64);
  const u16* Kb = Kf + (size_t)b * KVLEN * DM + h * DH;
  const u16* Vb = VT + ((size_t)b * NH + h) * (size_t)DH * KVLEN;

  for (int kv0 = 0; kv0 < KVLEN; kv0 += 64) {
    // ---- S = q @ K^T over a 64-kv tile (Bfrag rows = K_full rows, K-dim = dh)
    v8f s[4] = {};
#pragma unroll
    for (int t = 0; t < 4; ++t) {
      const u16* kr = Kb + (size_t)(kv0 + t * 16 + ln16) * DM;
      FragAB b0 = load_frag(kr, 0, hh);
      FragAB b1 = load_frag(kr, 32, hh);
      s[t] = wmma_bf16(aq0.v, b0.v, s[t]);
      s[t] = wmma_bf16(aq1.v, b1.v, s[t]);
    }
    // ---- online softmax over rows (row stats replicated across 16-lane halves)
    float mnew[8], fac[8], rs[8];
#pragma unroll
    for (int v = 0; v < 8; ++v) {
      float mx = fmaxf(fmaxf(s[0][v], s[1][v]), fmaxf(s[2][v], s[3][v]));
      mx = fmaxf(mx, __shfl_xor(mx, 1, 32));
      mx = fmaxf(mx, __shfl_xor(mx, 2, 32));
      mx = fmaxf(mx, __shfl_xor(mx, 4, 32));
      mx = fmaxf(mx, __shfl_xor(mx, 8, 32));
      mnew[v] = fmaxf(mi[v], mx);
      fac[v]  = __expf(mi[v] - mnew[v]);
      rs[v]   = 0.0f;
    }
    __syncthreads();
#pragma unroll
    for (int t = 0; t < 4; ++t) {
#pragma unroll
      for (int v = 0; v < 8; ++v) {
        float p = __expf(s[t][v] - mnew[v]);
        rs[v] += p;
        myP[(v + hh * 8) * 64 + t * 16 + ln16] = f2bf(p);   // C-layout -> row-major
      }
    }
#pragma unroll
    for (int v = 0; v < 8; ++v) {
      float r = rs[v];
      r += __shfl_xor(r, 1, 32);
      r += __shfl_xor(r, 2, 32);
      r += __shfl_xor(r, 4, 32);
      r += __shfl_xor(r, 8, 32);
      li[v] = li[v] * fac[v] + r;
      mi[v] = mnew[v];
    }
#pragma unroll
    for (int j = 0; j < 4; ++j)
#pragma unroll
      for (int v = 0; v < 8; ++v) acc[j][v] *= fac[v];
    __syncthreads();
    // ---- O += P @ V   (A = P from LDS; B^T = V^T rows, K-dim = kv)
    const u16* pr = myP + ln16 * 64;
    FragAB ap0 = load_frag(pr, 0, hh);
    FragAB ap1 = load_frag(pr, 32, hh);
#pragma unroll
    for (int j = 0; j < 4; ++j) {
      const u16* vr = Vb + (size_t)(j * 16 + ln16) * KVLEN + kv0;
      FragAB b0 = load_frag(vr, 0, hh);
      FragAB b1 = load_frag(vr, 32, hh);
      acc[j] = wmma_bf16(ap0.v, b0.v, acc[j]);
      acc[j] = wmma_bf16(ap1.v, b1.v, acc[j]);
    }
  }
#pragma unroll
  for (int j = 0; j < 4; ++j)
#pragma unroll
    for (int v = 0; v < 8; ++v) {
      float o = acc[j][v] / li[v];
      Obf[(size_t)(qrow0 + v + hh * 8) * DM + h * DH + j * 16 + ln16] = f2bf(o);
    }
}

// ---------------- residual + LayerNorm (writes fp32 state + bf16 copy) ----------
__global__ __launch_bounds__(256)
void add_ln(const float* __restrict__ resid, const float* __restrict__ delta,
            const float* __restrict__ gg, const float* __restrict__ bb,
            float* __restrict__ xout, u16* __restrict__ xbf) {
  __shared__ float s1[8], s2[8];
  const int row = blockIdx.x;
  const int tid = threadIdx.x;
  const int lane = tid & 31, wv = tid >> 5;
  float v4[4];
  float sum = 0.f, sq = 0.f;
  const size_t base = (size_t)row * DM + tid * 4;
#pragma unroll
  for (int i = 0; i < 4; ++i) {
    float v = resid[base + i] + delta[base + i];
    v4[i] = v; sum += v; sq += v * v;
  }
#pragma unroll
  for (int off = 16; off > 0; off >>= 1) {
    sum += __shfl_down(sum, off, 32);
    sq  += __shfl_down(sq, off, 32);
  }
  if (lane == 0) { s1[wv] = sum; s2[wv] = sq; }
  __syncthreads();
  float S = 0.f, Q = 0.f;
#pragma unroll
  for (int i = 0; i < 8; ++i) { S += s1[i]; Q += s2[i]; }
  const float mean = S * (1.0f / DM);
  const float var  = Q * (1.0f / DM) - mean * mean;
  const float rstd = rsqrtf(var + 1e-5f);
#pragma unroll
  for (int i = 0; i < 4; ++i) {
    int c = tid * 4 + i;
    float o = (v4[i] - mean) * rstd * gg[c] + bb[c];
    xout[base + i] = o;
    xbf[base + i]  = f2bf(o);
  }
}

// ---------------- host orchestration ----------------
extern "C" void kernel_launch(void* const* d_in, const int* in_sizes, int n_in,
                              void* d_out, int out_size, void* d_ws, size_t ws_size,
                              hipStream_t stream) {
  (void)in_sizes; (void)n_in; (void)out_size; (void)ws_size;
  const float* x          = (const float*)d_in[0];
  const int*   kv_indptr  = (const int*)d_in[2];
  const int*   kv_indices = (const int*)d_in[3];
  const float* pagetable  = (const float*)d_in[5];
  const float* Wq = (const float*)d_in[6];
  const float* bq = (const float*)d_in[7];
  const float* Wk = (const float*)d_in[8];
  const float* bk = (const float*)d_in[9];
  const float* Wv = (const float*)d_in[10];
  const float* bv = (const float*)d_in[11];
  const float* Wo = (const float*)d_in[12];
  const float* bo = (const float*)d_in[13];
  const float* ln1g = (const float*)d_in[14];
  const float* ln1b = (const float*)d_in[15];
  const float* W1 = (const float*)d_in[16];
  const float* b1 = (const float*)d_in[17];
  const float* W2 = (const float*)d_in[18];
  const float* b2 = (const float*)d_in[19];
  const float* ln2g = (const float*)d_in[20];
  const float* ln2b = (const float*)d_in[21];

  char* ws = (char*)d_ws;
  size_t off = 0;
  auto carve = [&](size_t bytes) -> void* {
    void* p = ws + off;
    off += (bytes + 255) & ~(size_t)255;
    return p;
  };
  u16*   wtq  = (u16*)carve((size_t)DM * DM * 2);
  u16*   wtk  = (u16*)carve((size_t)DM * DM * 2);
  u16*   wtv  = (u16*)carve((size_t)DM * DM * 2);
  u16*   wto  = (u16*)carve((size_t)DM * DM * 2);
  u16*   wt1  = (u16*)carve((size_t)DM * FF * 2);
  u16*   wt2  = (u16*)carve((size_t)FF * DM * 2);
  float* xcur = (float*)carve((size_t)NT * DM * 4);
  u16*   xbf  = (u16*)carve((size_t)NT * DM * 2);
  u16*   qbf  = (u16*)carve((size_t)NT * DM * 2);
  u16*   kbf  = (u16*)carve((size_t)NT * DM * 2);
  u16*   vbf  = (u16*)carve((size_t)NT * DM * 2);
  u16*   Kful = (u16*)carve((size_t)NB * KVLEN * DM * 2);
  u16*   VTf  = (u16*)carve((size_t)NB * KVLEN * DM * 2);
  u16*   obf  = (u16*)carve((size_t)NT * DM * 2);
  float* tmpF = (float*)carve((size_t)NT * DM * 4);
  u16*   hbf  = (u16*)carve((size_t)NT * FF * 2);

  cast_init<<<(NT * DM) / 256, 256, 0, stream>>>(x, xcur, xbf);

  const float scl = 0.125f;  // DH^-0.5
  for (int l = 0; l < NL; ++l) {
    transpose_cast<<<(DM * DM) / 256, 256, 0, stream>>>(Wq + (size_t)l * DM * DM, wtq, DM, DM);
    transpose_cast<<<(DM * DM) / 256, 256, 0, stream>>>(Wk + (size_t)l * DM * DM, wtk, DM, DM);
    transpose_cast<<<(DM * DM) / 256, 256, 0, stream>>>(Wv + (size_t)l * DM * DM, wtv, DM, DM);
    transpose_cast<<<(DM * DM) / 256, 256, 0, stream>>>(Wo + (size_t)l * DM * DM, wto, DM, DM);
    transpose_cast<<<(DM * FF) / 256, 256, 0, stream>>>(W1 + (size_t)l * DM * FF, wt1, DM, FF);
    transpose_cast<<<(FF * DM) / 256, 256, 0, stream>>>(W2 + (size_t)l * FF * DM, wt2, FF, DM);

    dim3 blk(256);
    dim3 gQ(DM / 64, NT / 64);
    gemm_wmma_bf16<1><<<gQ, blk, 0, stream>>>(xbf, wtq, bq + l * DM, nullptr, qbf, NT, DM, DM, scl);
    gemm_wmma_bf16<1><<<gQ, blk, 0, stream>>>(xbf, wtk, bk + l * DM, nullptr, kbf, NT, DM, DM, 1.f);
    gemm_wmma_bf16<1><<<gQ, blk, 0, stream>>>(xbf, wtv, bv + l * DM, nullptr, vbf, NT, DM, DM, 1.f);

    gather_kv<<<(NB * KVLEN * DM) / 256, blk, 0, stream>>>(pagetable, kv_indptr, kv_indices,
                                                           kbf, vbf, Kful, VTf, l);
    attn_flash<<<NB * NH * (SQ / 128), blk, 0, stream>>>(qbf, Kful, VTf, obf);

    gemm_wmma_bf16<0><<<gQ, blk, 0, stream>>>(obf, wto, bo + l * DM, tmpF, nullptr, NT, DM, DM, 1.f);
    add_ln<<<NT, blk, 0, stream>>>(xcur, tmpF, ln1g + l * DM, ln1b + l * DM, xcur, xbf);

    dim3 gF1(FF / 64, NT / 64);
    gemm_wmma_bf16<2><<<gF1, blk, 0, stream>>>(xbf, wt1, b1 + l * FF, nullptr, hbf, NT, FF, DM, 1.f);
    dim3 gF2(DM / 64, NT / 64);
    gemm_wmma_bf16<0><<<gF2, blk, 0, stream>>>(hbf, wt2, b2 + l * DM, tmpF, nullptr, NT, DM, FF, 1.f);
    add_ln<<<NT, blk, 0, stream>>>(xcur, tmpF, ln2g + l * DM, ln2b + l * DM, xcur, xbf);
  }

  hipMemcpyAsync(d_out, xcur, (size_t)NT * DM * 4, hipMemcpyDeviceToDevice, stream);
}
